// EncoderLayer_14001593385061
// MI455X (gfx1250) — matrix operand order repaired and checked by reading, not compiled, not run on hardware
//
#include <hip/hip_runtime.h>
#include <hip/hip_bf16.h>

// ---------------------------------------------------------------------------
// Types for WMMA fragments (wave32, V_WMMA_F32_16X16X32_F16)
// ---------------------------------------------------------------------------
typedef __attribute__((ext_vector_type(16))) _Float16 v16h;
typedef __attribute__((ext_vector_type(8)))  _Float16 v8h;
typedef __attribute__((ext_vector_type(4)))  _Float16 v4h;
typedef __attribute__((ext_vector_type(8)))  float    v8f;
typedef __attribute__((ext_vector_type(4)))  unsigned int u32x4;
typedef __attribute__((ext_vector_type(8)))  int          i32x8;
typedef __attribute__((ext_vector_type(4)))  int          i32x4;

union Frag {
    v16h f;
    v8h  h[2];
};

#define D_MODEL 1024
#define SEQ     2048
#define NBATCH  4
#define NHEAD   16
#define DHEAD   64
#define FFDIM   4096
#define NROWS   (NBATCH * SEQ)   // 8192 token rows

// ---------------------------------------------------------------------------
// Tensor Data Mover support (probe-verified builtins; arity differs between
// toolchains -> select via the TDM header that only therock ships).
// ---------------------------------------------------------------------------
#if __has_builtin(__builtin_amdgcn_tensor_load_to_lds) && \
    __has_builtin(__builtin_amdgcn_s_wait_tensorcnt)
#define USE_TDM 1
#else
#define USE_TDM 0
#endif

#if USE_TDM
// Async DMA of a 2D f16 tile [tile1 rows x tile0 cols] from global (row
// stride 'stride0' elements, dim0 contiguous) into LDS at byte address
// lds_byte.  LDS padding: 4 DWORDs after every 16 DWORDs, so a 32-half row
// becomes a 40-half (80 B, 16B-aligned) LDS row -> b128 fragment loads with
// bank-staggered rows.
__device__ __forceinline__ void tdm_tile_to_lds(const _Float16* gsrc,
                                                unsigned lds_byte,
                                                unsigned dim0, unsigned dim1,
                                                unsigned stride0,
                                                unsigned tile0, unsigned tile1) {
    unsigned long long ga = (unsigned long long)(const void*)gsrc;
    u32x4 g0;
    g0.x = 1u;                                              // count=1
    g0.y = lds_byte;                                        // lds_addr
    g0.z = (unsigned)ga;                                    // global_addr lo
    g0.w = ((unsigned)(ga >> 32) & 0x01ffffffu) | 0x80000000u;  // hi | type=2
    i32x8 g1;
    g1[0] = (1 << 16) | (1 << 20) | (3 << 22) | (3 << 25);  // 2B, pad 4dw/16dw
    g1[1] = (int)((dim0 & 0xffffu) << 16);                  // tensor_dim0 lo
    g1[2] = (int)((dim0 >> 16) | ((dim1 & 0xffffu) << 16)); // dim0 hi|dim1 lo
    g1[3] = (int)((dim1 >> 16) | (tile0 << 16));            // dim1 hi|tile0
    g1[4] = (int)tile1;                                     // tile1, tile2=0
    g1[5] = (int)stride0;                                   // dim0_stride lo
    g1[6] = 0;
    g1[7] = 0;
    i32x4 z4 = {0, 0, 0, 0};
#if __has_include(<hip/amd_detail/amd_gfx1250_TDM.h>)
    i32x8 z8 = {0, 0, 0, 0, 0, 0, 0, 0};                    // 6-arg toolchain
    __builtin_amdgcn_tensor_load_to_lds(g0, g1, z4, z4, z8, 0);
#else
    __builtin_amdgcn_tensor_load_to_lds(g0, g1, z4, z4, 0); // ROCm 7.2
#endif
}
#define LDS_BYTE_ADDR(p) ((unsigned)(unsigned long long)(const void*)(p))
#endif

// ---------------------------------------------------------------------------
// fp32 -> fp16 conversion (grid-stride)
// ---------------------------------------------------------------------------
__global__ void k_f32_to_f16(const float* __restrict__ s,
                             _Float16* __restrict__ d, int n) {
    for (int i = blockIdx.x * blockDim.x + threadIdx.x; i < n;
         i += gridDim.x * blockDim.x) {
        d[i] = (_Float16)s[i];
    }
}

// fp32 [K][N] row-major  ->  fp16 [N][K] (transposed).  Makes the GEMM
// B-operand k-contiguous so fragments are wide b128 loads everywhere.
__global__ void k_f32_to_f16_t(const float* __restrict__ s,
                               _Float16* __restrict__ d, int K, int N) {
    for (int i = blockIdx.x * blockDim.x + threadIdx.x; i < K * N;
         i += gridDim.x * blockDim.x) {
        int k = i / N;
        int n = i - k * N;
        d[(size_t)n * K + k] = (_Float16)s[i];
    }
}

// ---------------------------------------------------------------------------
// GEMM: out[M,N] = A[M,K](f16,row-major) @ Wt[N,K](f16, pre-transposed) + bias
// Block: 128 threads (4 waves). Macro-tile 64(M) x 64(N); wave w owns rows
// [w*16, w*16+16) x 64 cols (4 WMMA tiles, f32 accum).
// Wt k-slabs (64 n-rows x 32 k) are DMA'd into LDS by the TDM (dim0 = k is
// contiguous in Wt), double-buffered so the async copy of slab k+1 overlaps
// the WMMAs of slab k.  B fragments are two ds_load_b128 per tile.
// mode: 0 = f32 out, 1 = f16 out, 2 = f16 out + ReLU,
//       3 = f16 transposed out (out[n*M + m])
// ---------------------------------------------------------------------------
__global__ __launch_bounds__(128)
void k_gemm_f16(const _Float16* __restrict__ A, const _Float16* __restrict__ Wt,
                const float* __restrict__ bias, void* __restrict__ out,
                int M, int N, int K, int mode) {
    // [n][k] slabs, row stride 40 halves (TDM pad / manual pad).
    __shared__ _Float16 ldsW[USE_TDM ? 2 : 1][64 * 40];

    const int tid  = threadIdx.x;
    const int lane = tid & 31;
    const int wid  = tid >> 5;
    const int n16  = lane & 15;
    const int hi   = lane >> 4;
    const int hi8  = hi << 3;            // accumulator row offset
    const int hi16 = hi << 4;            // B-fragment k offset
    const int bn0  = blockIdx.x * 64;
    const int bm0  = blockIdx.y * 64;

    v8f acc[4] = {};
    const _Float16* arow = A + (size_t)(bm0 + wid * 16 + n16) * K;
    const int nsteps = K >> 5;

#if USE_TDM
    if (wid == 0)
        tdm_tile_to_lds(Wt + (size_t)bn0 * K, LDS_BYTE_ADDR(&ldsW[0][0]),
                        (unsigned)K, (unsigned)N, (unsigned)K, 32, 64);
#endif

    for (int i = 0; i < nsteps; ++i) {
        const int k0 = i << 5;
        __syncthreads();   // readers of the slab being overwritten are done
#if USE_TDM
        if (wid == 0) {
            if (i + 1 < nsteps) {
                tdm_tile_to_lds(Wt + (size_t)bn0 * K + k0 + 32,
                                LDS_BYTE_ADDR(&ldsW[(i + 1) & 1][0]),
                                (unsigned)K, (unsigned)N, (unsigned)K, 32, 64);
                __builtin_amdgcn_s_wait_tensorcnt(1);   // slab i landed
            } else {
                __builtin_amdgcn_s_wait_tensorcnt(0);
            }
        }
        const _Float16* wb = &ldsW[i & 1][0];
#else
        // Manual staging: rows of Wt are k-contiguous -> wide copies.
#pragma unroll
        for (int u = 0; u < 2; ++u) {
            int c   = tid + u * 128;
            int row = c >> 2;
            int off = (c & 3) << 3;
            *(v8h*)&ldsW[0][row * 40 + off] =
                *(const v8h*)(Wt + (size_t)(bn0 + row) * K + k0 + off);
        }
        if (k0 + 32 < K)
            __builtin_prefetch(Wt + (size_t)(bn0 + (tid >> 1)) * K + k0 + 32, 0, 0);
        const _Float16* wb = &ldsW[0][0];
#endif
        __syncthreads();

        Frag af;
        af.h[0] = *(const v8h*)(arow + k0 + hi16);
        af.h[1] = *(const v8h*)(arow + k0 + hi16 + 8);

        Frag bf[4];
#pragma unroll
        for (int t = 0; t < 4; ++t) {   // B layout: lane=col n, k=hi*16+e
            const _Float16* bp = wb + (t * 16 + n16) * 40;
            bf[t].h[0] = *(const v8h*)(bp + hi16);
            bf[t].h[1] = *(const v8h*)(bp + hi16 + 8);
        }
#pragma unroll
        for (int t = 0; t < 4; ++t)
            acc[t] = __builtin_amdgcn_wmma_f32_16x16x32_f16(
                false, af.f, false, bf[t].f, (short)0, acc[t], false, false);
    }

    // Epilogue: C/D layout -> lane holds col n16, rows r + 8*hi.
#pragma unroll
    for (int t = 0; t < 4; ++t) {
        int n = bn0 + t * 16 + n16;
        float bv = bias[n];
#pragma unroll
        for (int r = 0; r < 8; ++r) {
            int m = bm0 + wid * 16 + r + hi8;
            float val = acc[t][r] + bv;
            if (mode == 0) {
                ((float*)out)[(size_t)m * N + n] = val;
            } else if (mode == 3) {
                ((_Float16*)out)[(size_t)n * M + m] = (_Float16)val;
            } else {
                if (mode == 2 && val < 0.f) val = 0.f;
                ((_Float16*)out)[(size_t)m * N + n] = (_Float16)val;
            }
        }
    }
}

// ---------------------------------------------------------------------------
// Row reductions within each 16-lane half.
// ---------------------------------------------------------------------------
__device__ __forceinline__ float rmax16(float v) {
#pragma unroll
    for (int m = 1; m < 16; m <<= 1) v = fmaxf(v, __shfl_xor(v, m, 32));
    return v;
}
__device__ __forceinline__ float rsum16(float v) {
#pragma unroll
    for (int m = 1; m < 16; m <<= 1) v += __shfl_xor(v, m, 32);
    return v;
}

// ---------------------------------------------------------------------------
// Flash attention: one wave per (b, h, 16-query tile). Streams 32 keys per
// iteration: scores = Q(16x64) @ K^T(64x32) (4 WMMAs), online softmax in f32,
// O += P(16x32) @ V(32x64) (4 WMMAs).
// V is stored TRANSPOSED by the V-projection GEMM: Vt[h*64+dh][b*S + s]
// (row stride NROWS), so V B-fragments are direct global b128 loads and no
// V staging is needed.  Only P round-trips through (tiny) LDS.
// ---------------------------------------------------------------------------
__global__ __launch_bounds__(32)
void k_attn(const _Float16* __restrict__ Q, const _Float16* __restrict__ Km,
            const _Float16* __restrict__ Vt, _Float16* __restrict__ O) {
    __shared__ _Float16 pbuf[16 * 40];       // P[m][k], 16x32 (stride 40)

    const int lane = threadIdx.x;
    const int n16  = lane & 15;
    const int hi   = lane >> 4;
    const int hi8  = hi << 3;
    const int hi16 = hi << 4;
    const int q0 = blockIdx.x * 16;
    const int h  = blockIdx.y;
    const int b  = blockIdx.z;
    const size_t base = (size_t)b * SEQ * D_MODEL + h * DHEAD;

    Frag qf[2];
    {
        const _Float16* qrow = Q + base + (size_t)(q0 + n16) * D_MODEL;
#pragma unroll
        for (int g = 0; g < 2; ++g) {
            qf[g].h[0] = *(const v8h*)(qrow + g * 32 + hi16);
            qf[g].h[1] = *(const v8h*)(qrow + g * 32 + hi16 + 8);
        }
    }

    v8f o[4] = {};
    float mrow[8], lrow[8];
#pragma unroll
    for (int r = 0; r < 8; ++r) { mrow[r] = -1e30f; lrow[r] = 0.f; }

    for (int j = 0; j < SEQ / 32; ++j) {
        // ----- scores: two 16x16 tiles over this 32-key chunk -----
        v8f st[2];
#pragma unroll
        for (int t = 0; t < 2; ++t) {
            const _Float16* krow =
                Km + base + (size_t)(j * 32 + t * 16 + n16) * D_MODEL;
            v8f c = {};
#pragma unroll
            for (int g = 0; g < 2; ++g) {
                Frag kf;   // B-fragment: lane = key col, k contiguous
                kf.h[0] = *(const v8h*)(krow + g * 32 + hi16);
                kf.h[1] = *(const v8h*)(krow + g * 32 + hi16 + 8);
                c = __builtin_amdgcn_wmma_f32_16x16x32_f16(
                    false, qf[g].f, false, kf.f, (short)0, c, false, false);
            }
            st[t] = c;
        }

        __syncthreads();   // pbuf WAR vs previous iteration reads

        // ----- online softmax update (rows r + 8*hi, per-lane col n16) -----
        float corr[8];
#pragma unroll
        for (int r = 0; r < 8; ++r) {
            float s0 = st[0][r] * 0.125f;   // 1/sqrt(64)
            float s1 = st[1][r] * 0.125f;
            float mn = fmaxf(mrow[r], rmax16(fmaxf(s0, s1)));
            float c0 = __expf(mrow[r] - mn);
            float p0 = __expf(s0 - mn);
            float p1 = __expf(s1 - mn);
            lrow[r] = lrow[r] * c0 + rsum16(p0 + p1);
            mrow[r] = mn;
            corr[r] = c0;
            int m = r + hi8;
            pbuf[m * 40 + n16]      = (_Float16)p0;
            pbuf[m * 40 + 16 + n16] = (_Float16)p1;
        }
#pragma unroll
        for (int t = 0; t < 4; ++t)
#pragma unroll
            for (int r = 0; r < 8; ++r) o[t][r] *= corr[r];

        __syncthreads();

        // ----- O += P @ V  (V fragments straight from transposed global) ---
        Frag pf;   // A-fragment of P: lane = query row n16
        pf.h[0] = *(const v8h*)(&pbuf[n16 * 40 + hi16]);
        pf.h[1] = *(const v8h*)(&pbuf[n16 * 40 + hi16 + 8]);
#pragma unroll
        for (int t = 0; t < 4; ++t) {
            Frag vf;   // lane = dh col, keys contiguous in Vt
            const _Float16* vp = Vt +
                (size_t)(h * DHEAD + t * 16 + n16) * NROWS +
                (size_t)b * SEQ + j * 32;
            vf.h[0] = *(const v8h*)(vp + hi16);
            vf.h[1] = *(const v8h*)(vp + hi16 + 8);
            o[t] = __builtin_amdgcn_wmma_f32_16x16x32_f16(
                false, pf.f, false, vf.f, (short)0, o[t], false, false);
        }
    }

    // ----- finalize: divide by row sums, write f16 -----
#pragma unroll
    for (int t = 0; t < 4; ++t)
#pragma unroll
        for (int r = 0; r < 8; ++r) {
            int m = r + hi8;
            O[base + (size_t)(q0 + m) * D_MODEL + t * 16 + n16] =
                (_Float16)(o[t][r] / lrow[r]);
        }
}

// ---------------------------------------------------------------------------
// Fused residual + LayerNorm (torch semantics: unbiased std, eps on sigma).
// ---------------------------------------------------------------------------
__global__ __launch_bounds__(256)
void k_ln(const float* __restrict__ za, const float* __restrict__ zb,
          const float* __restrict__ ga, const float* __restrict__ be,
          float* __restrict__ of32, _Float16* __restrict__ of16, int write16) {
    __shared__ float red[256];
    const int row = blockIdx.x;
    const int tid = threadIdx.x;
    const size_t basei = (size_t)row * D_MODEL + tid * 4;

    float4 a = *(const float4*)(za + basei);
    float4 b = *(const float4*)(zb + basei);
    float z[4] = {a.x + b.x, a.y + b.y, a.z + b.z, a.w + b.w};

    red[tid] = z[0] + z[1] + z[2] + z[3];
    __syncthreads();
    for (int o = 128; o > 0; o >>= 1) {
        if (tid < o) red[tid] += red[tid + o];
        __syncthreads();
    }
    float mu = red[0] * (1.f / 1024.f);
    __syncthreads();

    float d0 = z[0] - mu, d1 = z[1] - mu, d2 = z[2] - mu, d3 = z[3] - mu;
    red[tid] = d0 * d0 + d1 * d1 + d2 * d2 + d3 * d3;
    __syncthreads();
    for (int o = 128; o > 0; o >>= 1) {
        if (tid < o) red[tid] += red[tid + o];
        __syncthreads();
    }
    float var = red[0] * (1.f / 1023.f);   // ddof = 1
    float inv = 1.f / (sqrtf(var) + 1e-3f);

    float4 g  = *(const float4*)(ga + tid * 4);
    float4 bb = *(const float4*)(be + tid * 4);
    float o0 = d0 * inv * g.x + bb.x;
    float o1 = d1 * inv * g.y + bb.y;
    float o2 = d2 * inv * g.z + bb.z;
    float o3 = d3 * inv * g.w + bb.w;
    *(float4*)(of32 + basei) = make_float4(o0, o1, o2, o3);
    if (write16) {
        v4h hv = {(_Float16)o0, (_Float16)o1, (_Float16)o2, (_Float16)o3};
        *(v4h*)(of16 + basei) = hv;
    }
}

// ---------------------------------------------------------------------------
// Launch
// ---------------------------------------------------------------------------
extern "C" void kernel_launch(void* const* d_in, const int* in_sizes, int n_in,
                              void* d_out, int out_size, void* d_ws,
                              size_t ws_size, hipStream_t stream) {
    (void)in_sizes; (void)n_in; (void)out_size; (void)ws_size;

    const float* x    = (const float*)d_in[0];
    const float* Wq   = (const float*)d_in[1];
    const float* bq   = (const float*)d_in[2];
    const float* Wk   = (const float*)d_in[3];
    const float* bk   = (const float*)d_in[4];
    const float* Wv   = (const float*)d_in[5];
    const float* bv   = (const float*)d_in[6];
    const float* Wo   = (const float*)d_in[7];
    const float* bo   = (const float*)d_in[8];
    const float* l1a  = (const float*)d_in[9];
    const float* l1b  = (const float*)d_in[10];
    const float* W1   = (const float*)d_in[11];
    const float* b1   = (const float*)d_in[12];
    const float* W2   = (const float*)d_in[13];
    const float* b2   = (const float*)d_in[14];
    const float* l2a  = (const float*)d_in[15];
    const float* l2b  = (const float*)d_in[16];
    float* out = (float*)d_out;

    // Workspace layout
    char* p = (char*)d_ws;
    size_t off = 0;
    auto alloc = [&](size_t bytes) {
        char* r = p + off;
        off += (bytes + 255) & ~(size_t)255;
        return r;
    };
    _Float16* xh    = (_Float16*)alloc((size_t)NROWS * D_MODEL * 2);
    _Float16* Wqt   = (_Float16*)alloc((size_t)D_MODEL * D_MODEL * 2);
    _Float16* Wkt   = (_Float16*)alloc((size_t)D_MODEL * D_MODEL * 2);
    _Float16* Wvt   = (_Float16*)alloc((size_t)D_MODEL * D_MODEL * 2);
    _Float16* Wot   = (_Float16*)alloc((size_t)D_MODEL * D_MODEL * 2);
    _Float16* W1t   = (_Float16*)alloc((size_t)D_MODEL * FFDIM * 2);
    _Float16* W2t   = (_Float16*)alloc((size_t)FFDIM * D_MODEL * 2);
    _Float16* Qh    = (_Float16*)alloc((size_t)NROWS * D_MODEL * 2);
    _Float16* Kh    = (_Float16*)alloc((size_t)NROWS * D_MODEL * 2);
    _Float16* Vt    = (_Float16*)alloc((size_t)NROWS * D_MODEL * 2); // [D][B*S]
    _Float16* attnh = (_Float16*)alloc((size_t)NROWS * D_MODEL * 2);
    float*    h1    = (float*)   alloc((size_t)NROWS * D_MODEL * 4);
    float*    h2    = (float*)   alloc((size_t)NROWS * D_MODEL * 4);
    _Float16* h2h   = (_Float16*)alloc((size_t)NROWS * D_MODEL * 2);
    _Float16* gh    = (_Float16*)alloc((size_t)NROWS * FFDIM * 2);
    float*    h3    = (float*)   alloc((size_t)NROWS * D_MODEL * 4);

    // 1) fp32 -> fp16 conversions (weights transposed to [N][K])
    dim3 cb(256);
    k_f32_to_f16<<<4096, cb, 0, stream>>>(x, xh, NROWS * D_MODEL);
    k_f32_to_f16_t<<<2048, cb, 0, stream>>>(Wq, Wqt, D_MODEL, D_MODEL);
    k_f32_to_f16_t<<<2048, cb, 0, stream>>>(Wk, Wkt, D_MODEL, D_MODEL);
    k_f32_to_f16_t<<<2048, cb, 0, stream>>>(Wv, Wvt, D_MODEL, D_MODEL);
    k_f32_to_f16_t<<<2048, cb, 0, stream>>>(Wo, Wot, D_MODEL, D_MODEL);
    k_f32_to_f16_t<<<4096, cb, 0, stream>>>(W1, W1t, D_MODEL, FFDIM);
    k_f32_to_f16_t<<<4096, cb, 0, stream>>>(W2, W2t, FFDIM, D_MODEL);

    // 2) Q/K/V projections (Q,K row-major f16; V transposed f16)
    dim3 gb(128);
    dim3 gQ(D_MODEL / 64, NROWS / 64);
    k_gemm_f16<<<gQ, gb, 0, stream>>>(xh, Wqt, bq, Qh, NROWS, D_MODEL, D_MODEL, 1);
    k_gemm_f16<<<gQ, gb, 0, stream>>>(xh, Wkt, bk, Kh, NROWS, D_MODEL, D_MODEL, 1);
    k_gemm_f16<<<gQ, gb, 0, stream>>>(xh, Wvt, bv, Vt, NROWS, D_MODEL, D_MODEL, 3);

    // 3) flash attention
    dim3 ga(SEQ / 16, NHEAD, NBATCH);
    k_attn<<<ga, dim3(32), 0, stream>>>(Qh, Kh, Vt, attnh);

    // 4) output projection (f32 out) + LN1 (residual with original x)
    k_gemm_f16<<<gQ, gb, 0, stream>>>(attnh, Wot, bo, h1, NROWS, D_MODEL, D_MODEL, 0);
    k_ln<<<NROWS, 256, 0, stream>>>(h1, x, l1a, l1b, h2, h2h, 1);

    // 5) FFN
    dim3 gF1(FFDIM / 64, NROWS / 64);
    k_gemm_f16<<<gF1, gb, 0, stream>>>(h2h, W1t, b1, gh, NROWS, FFDIM, D_MODEL, 2);
    k_gemm_f16<<<gQ, gb, 0, stream>>>(gh, W2t, b2, h3, NROWS, D_MODEL, FFDIM, 0);

    // 6) LN2 (residual with h2) -> final output
    k_ln<<<NROWS, 256, 0, stream>>>(h3, h2, l2a, l2b, out, h2h, 0);
}
